// GATRegressor_5849745457837
// MI455X (gfx1250) — matrix operand order repaired and checked by reading, not compiled, not run on hardware
//
#include <hip/hip_runtime.h>
#include <hip/hip_bf16.h>
#include <math.h>

typedef __attribute__((ext_vector_type(16))) _Float16 v16h;
typedef __attribute__((ext_vector_type(8)))  float    v8f;

#define NN   100000
#define EE   1600000
#define IND  256
#define HH   4
#define CC   64

// ---------------------------------------------------------------------------
// Pre-convert weights W[K x M] (f32 row-major) -> Bt[M x K] (f16), so the
// WMMA B operand is a contiguous 32B run per lane per k-step.
// ---------------------------------------------------------------------------
__global__ void convert_wt(const float* __restrict__ W, _Float16* __restrict__ Bt,
                           int K, int M) {
  int i = blockIdx.x * blockDim.x + threadIdx.x;  // over K*M
  if (i >= K * M) return;
  int k = i % K, c = i / K;
  Bt[(size_t)c * K + k] = (_Float16)W[(size_t)k * M + c];
}

// ---------------------------------------------------------------------------
// WMMA GEMM: C[n x M] = A[n x K] * Bt^T, A f32 row-major (cvt inline),
// Bt f16 [M x K]. One wave computes TPW 16x16 tiles sharing one A fetch.
// Block = 4 waves; block covers 64*TPW columns. n%16==0, K%32==0 required.
// ---------------------------------------------------------------------------
template <int TPW>
__global__ __launch_bounds__(128) void gemm_wmma(
    const float* __restrict__ A, const _Float16* __restrict__ Bt,
    float* __restrict__ C, int K, int M) {
  const int lane = threadIdx.x & 31;
  const int wv   = threadIdx.x >> 5;  // 0..3
  const int row0 = blockIdx.x << 4;   // 16-row tile
  const int colW = (blockIdx.y * 4 + wv) * 16 * TPW;
  const int kh   = lane >> 4;         // lane half (0/1)
  const int m16  = lane & 15;

  const float* arow = A + (size_t)(row0 + m16) * K;

  v8f acc[TPW];
#pragma unroll
  for (int t = 0; t < TPW; ++t) acc[t] = (v8f){};

  for (int k = 0; k < K; k += 32) {
    // A operand (contiguous k order used here, same as B): lane half kh holds K = kh*16 + [0..15], row = lane%16.
    v16h av;
    const float* ap0 = arow + k + kh * 16;
#pragma unroll
    for (int i = 0; i < 8; ++i) {
      av[i]     = (_Float16)ap0[i];
      av[8 + i] = (_Float16)ap0[8 + i];
    }
    // B operand (same contiguous k order as A): lane half kh holds K = kh*16 + [0..15], col = lane%16
    // -> contiguous 16 halves in Bt.
#pragma unroll
    for (int t = 0; t < TPW; ++t) {
      const _Float16* bp = Bt + (size_t)(colW + t * 16 + m16) * K + k + kh * 16;
      v16h bv = *(const v16h*)bp;
      acc[t] = __builtin_amdgcn_wmma_f32_16x16x32_f16(
          false, av, false, bv, (short)0, acc[t], false, false);
    }
  }
  // D f32 layout: VGPR r -> row = kh*8 + r, col = lane%16.
#pragma unroll
  for (int t = 0; t < TPW; ++t) {
    float* crow = C + (size_t)(row0 + kh * 8) * M + colW + t * 16 + m16;
#pragma unroll
    for (int r = 0; r < 8; ++r) crow[(size_t)r * M] = acc[t][r];
  }
}

// a_s[n,h] = sum_c h[n,h,c]*att_s[h,c]; a_d likewise. One thread per (n,h).
__global__ void att_scores(const float* __restrict__ h,
                           const float* __restrict__ att_s,
                           const float* __restrict__ att_d,
                           float* __restrict__ as_, float* __restrict__ ad_,
                           int n, int Hh) {
  int i = blockIdx.x * blockDim.x + threadIdx.x;
  if (i >= n * Hh) return;
  int hd = i % Hh;
  const float* hp = h + (size_t)i * CC;  // contiguous since i = node*Hh+hd
  const float* sp = att_s + hd * CC;
  const float* dp = att_d + hd * CC;
  float ss = 0.f, dd = 0.f;
#pragma unroll 8
  for (int c = 0; c < CC; ++c) {
    float v = hp[c];
    ss += v * sp[c];
    dd += v * dp[c];
  }
  as_[i] = ss;
  ad_[i] = dd;
}

// zero agg[nF], set m[nH] = -inf, den[nH] = 0
__global__ void init_layer(float* __restrict__ agg, float* __restrict__ m,
                           float* __restrict__ den, int nF, int nH) {
  int i = blockIdx.x * blockDim.x + threadIdx.x;
  if (i < nF) agg[i] = 0.f;
  if (i < nH) {
    m[i]   = -__builtin_inff();
    den[i] = 0.f;
  }
}

__device__ __forceinline__ void atomicMaxFloat(float* addr, float v) {
  if (v >= 0.f)
    atomicMax((int*)addr, __float_as_int(v));
  else
    atomicMin((unsigned int*)addr, __float_as_uint(v));
}

__device__ __forceinline__ float lrelu02(float e) {
  return e > 0.f ? e : 0.2f * e;
}

// Pass 1: segment max of e over dst. One thread per edge (self-loops appended).
__global__ void edge_max(const int* __restrict__ ei,
                         const float* __restrict__ as_,
                         const float* __restrict__ ad_,
                         float* __restrict__ m, int Hh) {
  int i = blockIdx.x * blockDim.x + threadIdx.x;
  if (i >= EE + NN) return;
  int s = (i < EE) ? ei[i] : i - EE;
  int d = (i < EE) ? ei[EE + i] : i - EE;
  for (int hd = 0; hd < Hh; ++hd) {
    float e = lrelu02(as_[s * Hh + hd] + ad_[d * Hh + hd]);
    atomicMaxFloat(&m[d * Hh + hd], e);
  }
}

// Pass 2: den[dst,h] += exp(e - m[dst,h])
__global__ void edge_expsum(const int* __restrict__ ei,
                            const float* __restrict__ as_,
                            const float* __restrict__ ad_,
                            const float* __restrict__ m,
                            float* __restrict__ den, int Hh) {
  int i = blockIdx.x * blockDim.x + threadIdx.x;
  if (i >= EE + NN) return;
  int s = (i < EE) ? ei[i] : i - EE;
  int d = (i < EE) ? ei[EE + i] : i - EE;
  for (int hd = 0; hd < Hh; ++hd) {
    float e  = lrelu02(as_[s * Hh + hd] + ad_[d * Hh + hd]);
    float ex = __expf(e - m[d * Hh + hd]);
    unsafeAtomicAdd(&den[d * Hh + hd], ex);
  }
}

// Pass 3: agg[dst, j] += h[src, j] * alpha(dst, j>>6). One wave per edge,
// lanes stripe features (F % 32 == 0, Cc == 64 in both layers).
__global__ __launch_bounds__(256) void edge_aggregate(
    const int* __restrict__ ei, const float* __restrict__ h,
    const float* __restrict__ as_, const float* __restrict__ ad_,
    const float* __restrict__ m, const float* __restrict__ den,
    float* __restrict__ agg, int Hh, int F) {
  int wave = (blockIdx.x * blockDim.x + threadIdx.x) >> 5;
  int lane = threadIdx.x & 31;
  if (wave >= EE + NN) return;
  int s = (wave < EE) ? ei[wave] : wave - EE;
  int d = (wave < EE) ? ei[EE + wave] : wave - EE;
  const float* hs = h + (size_t)s * F;
  float* ag       = agg + (size_t)d * F;
  for (int j = lane; j < F; j += 32) {
    int hd      = j >> 6;  // Cc == 64
    float e     = lrelu02(as_[s * Hh + hd] + ad_[d * Hh + hd]);
    float alpha = __expf(e - m[d * Hh + hd]) / den[d * Hh + hd];
    unsafeAtomicAdd(&ag[j], hs[j] * alpha);
  }
}

// out[i] = elu(agg[i] + bias[i % F])
__global__ void bias_elu(const float* __restrict__ agg,
                         const float* __restrict__ bias,
                         float* __restrict__ out, int total, int F) {
  int i = blockIdx.x * blockDim.x + threadIdx.x;
  if (i >= total) return;
  float v = agg[i] + bias[i % F];
  out[i]  = v > 0.f ? v : (__expf(v) - 1.f);
}

// y[n] = h2[n,:]·lin_w + lin_b
__global__ void final_linear(const float* __restrict__ h2,
                             const float* __restrict__ lw,
                             const float* __restrict__ lb,
                             float* __restrict__ y) {
  int n = blockIdx.x * blockDim.x + threadIdx.x;
  if (n >= NN) return;
  const float* hp = h2 + (size_t)n * CC;
  float acc = lb[0];
#pragma unroll 8
  for (int c = 0; c < CC; ++c) acc += hp[c] * lw[c];
  y[n] = acc;
}

extern "C" void kernel_launch(void* const* d_in, const int* in_sizes, int n_in,
                              void* d_out, int out_size, void* d_ws, size_t ws_size,
                              hipStream_t stream) {
  const float* x      = (const float*)d_in[0];
  const int*   ei     = (const int*)d_in[1];
  const float* W1     = (const float*)d_in[2];
  const float* att_s1 = (const float*)d_in[3];
  const float* att_d1 = (const float*)d_in[4];
  const float* b1     = (const float*)d_in[5];
  const float* W2     = (const float*)d_in[6];
  const float* att_s2 = (const float*)d_in[7];
  const float* att_d2 = (const float*)d_in[8];
  const float* b2     = (const float*)d_in[9];
  const float* lin_w  = (const float*)d_in[10];
  const float* lin_b  = (const float*)d_in[11];
  float* out = (float*)d_out;

  const int F1 = HH * CC;  // 256
  const int F2 = CC;       // 64

  float* ws   = (float*)d_ws;
  float* h1   = ws;                           // [N, 256]  (later reused: hpost1)
  float* agg1 = h1 + (size_t)NN * F1;         // [N, 256]  (later reused below)
  float* as1  = agg1 + (size_t)NN * F1;       // [N, 4]
  float* ad1  = as1 + (size_t)NN * HH;        // [N, 4]
  float* m1   = ad1 + (size_t)NN * HH;        // [N, 4]
  float* d1   = m1 + (size_t)NN * HH;         // [N, 4]
  float* as2  = d1 + (size_t)NN * HH;         // [N]
  float* ad2  = as2 + (size_t)NN;             // [N]
  float* m2   = ad2 + (size_t)NN;             // [N]
  float* d2   = m2 + (size_t)NN;              // [N]
  _Float16* W1t = (_Float16*)(d2 + NN);       // [256, 256] f16 (W1 transposed)
  _Float16* W2t = W1t + (size_t)IND * F1;     // [64, 256]  f16 (W2 transposed)
  // layer-2 big buffers reuse the (dead) agg1 region: 3*N*64 < N*256
  float* h2     = agg1;                       // [N, 64]
  float* agg2   = h2 + (size_t)NN * F2;       // [N, 64]
  float* hpost2 = agg2 + (size_t)NN * F2;     // [N, 64]

  const int ETOT = EE + NN;
  const int TPB  = 256;

  // ---- One-shot weight conversion (tiny) ----
  convert_wt<<<(IND * F1 + TPB - 1) / TPB, TPB, 0, stream>>>(W1, W1t, IND, F1);
  convert_wt<<<(F1 * F2 + TPB - 1) / TPB, TPB, 0, stream>>>(W2, W2t, F1, F2);

  // ---- Layer 1 ----  (TPW=4: one block covers all 256 cols; x streamed once)
  gemm_wmma<4><<<dim3(NN / 16, 1), 128, 0, stream>>>(x, W1t, h1, IND, F1);
  att_scores<<<(NN * HH + TPB - 1) / TPB, TPB, 0, stream>>>(h1, att_s1, att_d1,
                                                            as1, ad1, NN, HH);
  init_layer<<<((size_t)NN * F1 + TPB - 1) / TPB, TPB, 0, stream>>>(
      agg1, m1, d1, NN * F1, NN * HH);
  edge_max<<<(ETOT + TPB - 1) / TPB, TPB, 0, stream>>>(ei, as1, ad1, m1, HH);
  edge_expsum<<<(ETOT + TPB - 1) / TPB, TPB, 0, stream>>>(ei, as1, ad1, m1, d1, HH);
  edge_aggregate<<<((size_t)ETOT * 32 + TPB - 1) / TPB, TPB, 0, stream>>>(
      ei, h1, as1, ad1, m1, d1, agg1, HH, F1);
  bias_elu<<<((size_t)NN * F1 + TPB - 1) / TPB, TPB, 0, stream>>>(
      agg1, b1, h1, NN * F1, F1);  // hpost1 -> h1 buffer

  // ---- Layer 2 ----  (M=64: TPW=1, one block covers 64 cols)
  gemm_wmma<1><<<dim3(NN / 16, 1), 128, 0, stream>>>(h1, W2t, h2, F1, F2);
  att_scores<<<(NN + TPB - 1) / TPB, TPB, 0, stream>>>(h2, att_s2, att_d2,
                                                       as2, ad2, NN, 1);
  init_layer<<<((size_t)NN * F2 + TPB - 1) / TPB, TPB, 0, stream>>>(
      agg2, m2, d2, NN * F2, NN);
  edge_max<<<(ETOT + TPB - 1) / TPB, TPB, 0, stream>>>(ei, as2, ad2, m2, 1);
  edge_expsum<<<(ETOT + TPB - 1) / TPB, TPB, 0, stream>>>(ei, as2, ad2, m2, d2, 1);
  edge_aggregate<<<((size_t)ETOT * 32 + TPB - 1) / TPB, TPB, 0, stream>>>(
      ei, h2, as2, ad2, m2, d2, agg2, 1, F2);
  bias_elu<<<((size_t)NN * F2 + TPB - 1) / TPB, TPB, 0, stream>>>(
      agg2, b2, hpost2, NN * F2, F2);

  // ---- Head ----
  final_linear<<<(NN + TPB - 1) / TPB, TPB, 0, stream>>>(hpost2, lin_w, lin_b, out);
}
